// GATEncoder_23725399343291
// MI455X (gfx1250) — matrix operand order repaired and checked by reading, not compiled, not run on hardware
//
#include <hip/hip_runtime.h>
#include <hip/hip_bf16.h>
#include <math.h>

// ---------------------------------------------------------------------------
// Types for CDNA5 WMMA (wave32, 16x16x32 bf16 -> f32 accumulate)
// ---------------------------------------------------------------------------
typedef __bf16 bf16_t;
typedef __attribute__((ext_vector_type(16))) __bf16 v16bf;
typedef __attribute__((ext_vector_type(8)))  float  v8f;

#define NEG_SLOPE 0.2f

__device__ __forceinline__ bf16_t f32_to_bf16(float f) {
  union { float f; unsigned u; } v; v.f = f;
  unsigned r = v.u + 0x7fffu + ((v.u >> 16) & 1u);   // round-to-nearest-even
  union { unsigned short u; bf16_t h; } o; o.u = (unsigned short)(r >> 16);
  return o.h;
}

// float atomic max via signed-max (val>=0) / unsigned-min (val<0) trick.
__device__ __forceinline__ void atomicMaxF(float* addr, float val) {
  if (val >= 0.0f) atomicMax((int*)addr, __float_as_int(val));
  else             atomicMin((unsigned int*)addr, (unsigned int)__float_as_int(val));
}

// ---------------------------------------------------------------------------
// 0) generic fill
// ---------------------------------------------------------------------------
__global__ void k_fill_f32(float* __restrict__ p, float v, int n) {
  int i = blockIdx.x * blockDim.x + threadIdx.x;
  if (i < n) p[i] = v;
}

// ---------------------------------------------------------------------------
// 1) normalize node embeddings, gather to x (fp32 + bf16)
//    emb = node_emb * min(1, 1/max(||row||, 1e-7)) ; x = emb[node_ids[:,0]]
//    one block (128 threads) per node
// ---------------------------------------------------------------------------
__global__ __launch_bounds__(128) void k_prep_nodes(
    const int* __restrict__ node_ids, const float* __restrict__ node_emb,
    float* __restrict__ x, bf16_t* __restrict__ xh, int D) {
  int n = blockIdx.x;
  int d = threadIdx.x;
  int id = node_ids[n];
  const float* row = node_emb + (size_t)id * D;
  float v = row[d];
  __shared__ float red[128];
  red[d] = v * v;
  __syncthreads();
  for (int s = 64; s > 0; s >>= 1) {
    if (d < s) red[d] += red[d + s];
    __syncthreads();
  }
  float nrm = sqrtf(red[0]);
  float scale = fminf(1.0f, 1.0f / fmaxf(nrm, 1e-7f));
  float o = v * scale;
  x [(size_t)n * D + d] = o;
  xh[(size_t)n * D + d] = f32_to_bf16(o);
}

// ---------------------------------------------------------------------------
// 2) W (K x Ncols, row-major fp32) -> Wt (Ncols x K, bf16) so B-tiles are
//    contiguous per lane in the WMMA GEMM.
// ---------------------------------------------------------------------------
__global__ void k_transpose_bf16(const float* __restrict__ W, bf16_t* __restrict__ Wt,
                                 int K, int Ncols) {
  int i = blockIdx.x * blockDim.x + threadIdx.x;
  if (i >= K * Ncols) return;
  int k = i / Ncols, c = i % Ncols;
  Wt[(size_t)c * K + k] = f32_to_bf16(W[i]);
}

// ---------------------------------------------------------------------------
// 3) s_e[e,h] = ea[e] . att[h, 2D:2D+De]   (0 for self-loop edges e >= E)
// ---------------------------------------------------------------------------
__global__ void k_edge_se(const int* __restrict__ etype, const float* __restrict__ edge_emb,
                          const float* __restrict__ att, float* __restrict__ s_e,
                          int E, int Etot, int H, int D, int De, int astride) {
  int i = blockIdx.x * blockDim.x + threadIdx.x;
  if (i >= Etot * H) return;
  int e = i / H, h = i % H;
  float s = 0.0f;
  if (e < E) {
    const float* ea = edge_emb + (size_t)etype[e] * De;
    const float* aE = att + (size_t)h * astride + 2 * D;
    for (int d = 0; d < De; ++d) s += ea[d] * aE[d];
  }
  s_e[i] = s;
}

// ---------------------------------------------------------------------------
// 4) GEMM: C(MxN) = A(MxK, bf16 rowmajor) * B(KxN) with B given transposed
//    (Bt: N x K, bf16 rowmajor). One wave computes one 16x16 C-tile using
//    v_wmma_f32_16x16x32_bf16. Lane layouts per ISA 7.12.2.
//    GEMM core shared by a compile-time-K kernel (full unroll: 4 WMMAs
//    back-to-back, all loads clause-able) and a runtime-K fallback.
// ---------------------------------------------------------------------------
__device__ __forceinline__ void gemm_tile_body(
    const bf16_t* __restrict__ A, const bf16_t* __restrict__ Bt,
    float* __restrict__ C, int M, int K, int Ncols, bool unroll_hint) {
  const int tm   = blockIdx.x;
  const int tn   = blockIdx.y;
  const int lane = threadIdx.x;
  const int half = lane >> 4;   // 0: lanes 0-15, 1: lanes 16-31
  const int l16  = lane & 15;

  const int arow = tm * 16 + l16;
  const int bcol = tn * 16 + l16;
  const bool aok = (arow < M);
  const bf16_t* ap = A  + (size_t)(aok ? arow : 0) * K;
  const bf16_t* bp = Bt + (size_t)bcol * K;

  v8f acc = {};
  (void)unroll_hint;
  for (int k0 = 0; k0 < K; k0 += 32) {
    v16bf a, b;
    // A 16x32: elems 0..7 -> K = k0 + half*8 + i ; elems 8..15 -> +16
#pragma unroll
    for (int i = 0; i < 8; ++i) {
      a[i]     = ap[k0 + half * 8 + i];
      a[8 + i] = ap[k0 + 16 + half * 8 + i];
    }
    // B 32x16: elems 0..7 -> K = k0 + half*8 + i ; elems 8..15 -> +16 (column l16)
#pragma unroll
    for (int v = 0; v < 8; ++v) {
      b[v]     = bp[k0 + half * 8 + v];
      b[8 + v] = bp[k0 + 16 + half * 8 + v];
    }
    acc = __builtin_amdgcn_wmma_f32_16x16x32_bf16(
        /*neg_a=*/false, a, /*neg_b=*/false, b,
        /*c_mod=*/(short)0, acc, /*reuse_a=*/false, /*reuse_b=*/false);
  }

  // C/D layout: VGPR r -> row = r + half*8, col = l16.
  // Single tile-level bounds branch; full tiles take unconditional stores.
  const int row0 = tm * 16;
  float* cp = C + (size_t)(row0 + half * 8) * Ncols + tn * 16 + l16;
  if (row0 + 16 <= M) {
#pragma unroll
    for (int r = 0; r < 8; ++r) cp[(size_t)r * Ncols] = acc[r];
  } else {
#pragma unroll
    for (int r = 0; r < 8; ++r) {
      if (row0 + half * 8 + r < M) cp[(size_t)r * Ncols] = acc[r];
    }
  }
}

template <int K>
__global__ __launch_bounds__(32) void k_gemm_wmma_bf16_k(
    const bf16_t* __restrict__ A, const bf16_t* __restrict__ Bt,
    float* __restrict__ C, int M, int Ncols) {
  gemm_tile_body(A, Bt, C, M, K, Ncols, true);   // K is a compile-time constant
}

__global__ __launch_bounds__(32) void k_gemm_wmma_bf16_rt(
    const bf16_t* __restrict__ A, const bf16_t* __restrict__ Bt,
    float* __restrict__ C, int M, int K, int Ncols) {
  gemm_tile_body(A, Bt, C, M, K, Ncols, false);
}

// ---------------------------------------------------------------------------
// 5) s_i[n,h] = h[n,h,:] . att[h,:D] ; s_j[n,h] = h[n,h,:] . att[h,D:2D]
// ---------------------------------------------------------------------------
__global__ void k_node_scores(const float* __restrict__ hmat, const float* __restrict__ att,
                              float* __restrict__ s_i, float* __restrict__ s_j,
                              int N, int H, int D, int astride) {
  int i = blockIdx.x * blockDim.x + threadIdx.x;
  if (i >= N * H) return;
  int n = i / H, h = i % H;
  const float* hv = hmat + ((size_t)n * H + h) * D;
  const float* ai = att + (size_t)h * astride;
  const float* aj = ai + D;
  float si = 0.0f, sj = 0.0f;
  for (int d = 0; d < D; ++d) { si += hv[d] * ai[d]; sj += hv[d] * aj[d]; }
  s_i[i] = si;
  s_j[i] = sj;
}

// ---------------------------------------------------------------------------
// 6) pass 1: score = leaky_relu(s_i[dst] + s_j[src] + s_e); m[dst] = max
// ---------------------------------------------------------------------------
__global__ void k_edge_pass1(const int* __restrict__ src, const int* __restrict__ dst,
                             const float* __restrict__ s_i, const float* __restrict__ s_j,
                             const float* __restrict__ s_e, float* __restrict__ sc,
                             float* __restrict__ m, int E, int Etot, int H) {
  int i = blockIdx.x * blockDim.x + threadIdx.x;
  if (i >= Etot * H) return;
  int e = i / H, h = i % H;
  int s = (e < E) ? src[e] : (e - E);
  int t = (e < E) ? dst[e] : (e - E);
  float v = s_i[(size_t)t * H + h] + s_j[(size_t)s * H + h] + s_e[i];
  v = (v > 0.0f) ? v : NEG_SLOPE * v;
  sc[i] = v;
  atomicMaxF(&m[(size_t)t * H + h], v);
}

// ---------------------------------------------------------------------------
// 7) pass 2: e = exp(score - m[dst]); z[dst] += e
// ---------------------------------------------------------------------------
__global__ void k_edge_pass2(const int* __restrict__ dst, float* __restrict__ sc,
                             const float* __restrict__ m, float* __restrict__ z,
                             int E, int Etot, int H) {
  int i = blockIdx.x * blockDim.x + threadIdx.x;
  if (i >= Etot * H) return;
  int e = i / H, h = i % H;
  int t = (e < E) ? dst[e] : (e - E);
  float ev = __expf(sc[i] - m[(size_t)t * H + h]);
  sc[i] = ev;
  atomicAdd(&z[(size_t)t * H + h], ev);
}

// ---------------------------------------------------------------------------
// 8) pass 3: out[dst,d] += (1/H) * sum_h alpha[e,h] * h[src,h,d]
//    one wave per edge; head-mean folded in before the atomics.
// ---------------------------------------------------------------------------
__global__ __launch_bounds__(256) void k_edge_pass3(
    const int* __restrict__ src, const int* __restrict__ dst,
    const float* __restrict__ sc, const float* __restrict__ z,
    const float* __restrict__ hmat, float* __restrict__ out,
    int E, int Etot, int H, int D) {
  int wave = blockIdx.x * (blockDim.x >> 5) + (threadIdx.x >> 5);
  int lane = threadIdx.x & 31;
  if (wave >= Etot) return;
  int e = wave;
  int s = (e < E) ? src[e] : (e - E);
  int t = (e < E) ? dst[e] : (e - E);
  float invH = 1.0f / (float)H;
  float alpha[4];
  for (int h = 0; h < H; ++h)
    alpha[h] = sc[(size_t)e * H + h] / z[(size_t)t * H + h] * invH;
  const float* hs = hmat + (size_t)s * H * D;
  float*       op = out  + (size_t)t * D;
  for (int d = lane; d < D; d += 32) {
    float c = 0.0f;
    for (int h = 0; h < H; ++h) c += alpha[h] * hs[h * D + d];
    atomicAdd(&op[d], c);
  }
}

// ---------------------------------------------------------------------------
// 9) finalize: x = acc + b (; elu) ; also refresh bf16 copy for next GEMM
// ---------------------------------------------------------------------------
__global__ void k_finalize(const float* __restrict__ acc, const float* __restrict__ b,
                           float* __restrict__ xdst, bf16_t* __restrict__ xh,
                           int N, int D, int do_elu) {
  int i = blockIdx.x * blockDim.x + threadIdx.x;
  if (i >= N * D) return;
  float v = acc[i] + b[i % D];
  if (do_elu) v = (v > 0.0f) ? v : (__expf(v) - 1.0f);
  xdst[i] = v;
  xh[i] = f32_to_bf16(v);
}

// ---------------------------------------------------------------------------
// host
// ---------------------------------------------------------------------------
static inline void launch_gemm(const bf16_t* A, const bf16_t* Bt, float* C,
                               int M, int K, int Ncols, hipStream_t stream) {
  dim3 g((M + 15) / 16, (Ncols + 15) / 16);
  if (K == 128) k_gemm_wmma_bf16_k<128><<<g, 32, 0, stream>>>(A, Bt, C, M, Ncols);
  else          k_gemm_wmma_bf16_rt<<<g, 32, 0, stream>>>(A, Bt, C, M, K, Ncols);
}

extern "C" void kernel_launch(void* const* d_in, const int* in_sizes, int n_in,
                              void* d_out, int out_size, void* d_ws, size_t ws_size,
                              hipStream_t stream) {
  (void)n_in; (void)out_size; (void)ws_size;

  const int*   node_ids = (const int*)d_in[0];
  const int*   eidx     = (const int*)d_in[1];   // row0 = src, row1 = dst
  const int*   etype    = (const int*)d_in[2];
  const float* node_emb = (const float*)d_in[4];
  const float* edge_emb = (const float*)d_in[5];
  const float* W1   = (const float*)d_in[6];
  const float* att1 = (const float*)d_in[7];
  const float* b1   = (const float*)d_in[8];
  const float* W2   = (const float*)d_in[9];
  const float* att2 = (const float*)d_in[10];
  const float* b2   = (const float*)d_in[11];

  const int N  = in_sizes[0];          // node_ids is (N,1)
  const int E  = in_sizes[1] / 2;
  const int D  = in_sizes[8];          // b1 length
  const int HD = in_sizes[6] / D;      // W1 is D x (H*D)
  const int H  = HD / D;
  const int astride = in_sizes[7] / H; // 2D + De
  const int De = astride - 2 * D;
  const int Etot = E + N;

  const int* src = eidx;
  const int* dst = eidx + E;

  // ---- carve workspace (256B aligned) ----
  char* base = (char*)d_ws;
  size_t off = 0;
  auto carve = [&](size_t bytes) -> char* {
    char* p = base + off;
    off += (bytes + 255) & ~(size_t)255;
    return p;
  };
  float*  x    = (float*) carve((size_t)N * D * 4);
  bf16_t* xh   = (bf16_t*)carve((size_t)N * D * 2);
  bf16_t* W1t  = (bf16_t*)carve((size_t)HD * D * 2);       // (H*D) x D
  bf16_t* W2t  = (bf16_t*)carve((size_t)D * D * 2);        // D x D
  float*  hbuf = (float*) carve((size_t)N * HD * 4);       // N x (H*D)
  float*  s_i  = (float*) carve((size_t)N * H * 4);
  float*  s_j  = (float*) carve((size_t)N * H * 4);
  float*  s_e1 = (float*) carve((size_t)Etot * H * 4);
  float*  s_e2 = (float*) carve((size_t)Etot * 4);
  float*  mbuf = (float*) carve((size_t)N * H * 4);
  float*  zbuf = (float*) carve((size_t)N * H * 4);
  float*  sc   = (float*) carve((size_t)Etot * H * 4);
  float*  accu = (float*) carve((size_t)N * D * 4);

  const int TB = 256;
  auto nb = [&](long long n) { return (int)((n + TB - 1) / TB); };

  // ---- one-time prep ----
  k_prep_nodes<<<N, 128, 0, stream>>>(node_ids, node_emb, x, xh, D);
  k_transpose_bf16<<<nb((long long)D * HD), TB, 0, stream>>>(W1, W1t, D, HD);
  k_transpose_bf16<<<nb((long long)D * D),  TB, 0, stream>>>(W2, W2t, D, D);
  k_edge_se<<<nb((long long)Etot * H), TB, 0, stream>>>(etype, edge_emb, att1, s_e1,
                                                        E, Etot, H, D, De, astride);
  k_edge_se<<<nb((long long)Etot),     TB, 0, stream>>>(etype, edge_emb, att2, s_e2,
                                                        E, Etot, 1, D, De, astride);

  const int ROUNDS = 3;
  for (int r = 0; r < ROUNDS; ++r) {
    // ======== layer 1 (H heads) ========
    k_fill_f32<<<nb((long long)N * H), TB, 0, stream>>>(mbuf, -INFINITY, N * H);
    k_fill_f32<<<nb((long long)N * H), TB, 0, stream>>>(zbuf, 0.0f, N * H);
    k_fill_f32<<<nb((long long)N * D), TB, 0, stream>>>(accu, 0.0f, N * D);

    launch_gemm(xh, W1t, hbuf, N, D, HD, stream);
    k_node_scores<<<nb((long long)N * H), TB, 0, stream>>>(hbuf, att1, s_i, s_j, N, H, D, astride);
    k_edge_pass1<<<nb((long long)Etot * H), TB, 0, stream>>>(src, dst, s_i, s_j, s_e1, sc, mbuf, E, Etot, H);
    k_edge_pass2<<<nb((long long)Etot * H), TB, 0, stream>>>(dst, sc, mbuf, zbuf, E, Etot, H);
    k_edge_pass3<<<(Etot + 7) / 8, 256, 0, stream>>>(src, dst, sc, zbuf, hbuf, accu, E, Etot, H, D);
    k_finalize<<<nb((long long)N * D), TB, 0, stream>>>(accu, b1, x, xh, N, D, /*elu=*/1);

    // ======== layer 2 (1 head) ========
    k_fill_f32<<<nb((long long)N), TB, 0, stream>>>(mbuf, -INFINITY, N);
    k_fill_f32<<<nb((long long)N), TB, 0, stream>>>(zbuf, 0.0f, N);
    k_fill_f32<<<nb((long long)N * D), TB, 0, stream>>>(accu, 0.0f, N * D);

    launch_gemm(xh, W2t, hbuf, N, D, D, stream);
    k_node_scores<<<nb((long long)N), TB, 0, stream>>>(hbuf, att2, s_i, s_j, N, 1, D, astride);
    k_edge_pass1<<<nb((long long)Etot), TB, 0, stream>>>(src, dst, s_i, s_j, s_e2, sc, mbuf, E, Etot, 1);
    k_edge_pass2<<<nb((long long)Etot), TB, 0, stream>>>(dst, sc, mbuf, zbuf, E, Etot, 1);
    k_edge_pass3<<<(Etot + 7) / 8, 256, 0, stream>>>(src, dst, sc, zbuf, hbuf, accu, E, Etot, 1, D);

    float* xdst = (r == ROUNDS - 1) ? (float*)d_out : x;
    k_finalize<<<nb((long long)N * D), TB, 0, stream>>>(accu, b2, xdst, xh, N, D, /*elu=*/0);
  }
}
